// MultiHeadAttention_14173392077533
// MI455X (gfx1250) — matrix operand order repaired and checked by reading, not compiled, not run on hardware
//
#include <hip/hip_runtime.h>

// ---------------------------------------------------------------------------
// MI455X (gfx1250, wave32) multi-head attention:
//   qkv = x @ Wqkv + b ; q,k = mh_rms_norm(q,k) ; flash-attn ; out = h @ Wout + b
// GEMMs via v_wmma_f32_16x16x32_bf16 (fp32 accum); b128 LDS traffic;
// Q/K tiles via global_load_async_to_lds_b128 (ASYNCcnt); fixed-max softmax
// (RMS-normed q,k bound |score/sqrt(D)| <= 8) -> no running-max/rescale.
// ---------------------------------------------------------------------------

typedef __attribute__((ext_vector_type(16))) __bf16 v16bf;
typedef __attribute__((ext_vector_type(8)))  __bf16 v8bf;
typedef __attribute__((ext_vector_type(4)))  __bf16 v4bf;
typedef __attribute__((ext_vector_type(2)))  __bf16 v2bf;
typedef __attribute__((ext_vector_type(8)))  float  v8f;

#define Bn 4
#define Ln 2048
#define Cn 1024
#define Hn 16
#define Dn 64

__device__ __forceinline__ v8f wmma_bf16(v16bf a, v16bf b, v8f c) {
  return __builtin_amdgcn_wmma_f32_16x16x32_bf16(
      false, a, false, b, (short)0, c, false, false);
}

__device__ __forceinline__ v16bf cat16(v8bf lo, v8bf hi) {
  return __builtin_shufflevector(lo, hi, 0, 1, 2, 3, 4, 5, 6, 7,
                                 8, 9, 10, 11, 12, 13, 14, 15);
}

// A fragment (16x32 bf16) from row-major tile: two contiguous 16B runs/lane.
__device__ __forceinline__ v16bf load_a_frag(const __bf16* s, int m0, int k0, int ld) {
  const int lane = threadIdx.x & 31;
  const int half = lane >> 4, r = lane & 15;
  const __bf16* p = s + (m0 + r) * ld + k0 + half * 8;
  v8bf lo = *(const v8bf*)p;         // k = half*8 .. +7
  v8bf hi = *(const v8bf*)(p + 16);  // k = 16+half*8 .. +7
  return cat16(lo, hi);
}

// B fragment (32x16) from row-major (N x K) tile: two contiguous 16B runs/lane.
__device__ __forceinline__ v16bf load_b_frag_nk(const __bf16* s, int n0, int k0, int ld) {
  const int lane = threadIdx.x & 31;
  const int half = lane >> 4, r = lane & 15;
  const __bf16* p = s + (n0 + r) * ld + k0 + half * 8;
  v8bf lo = *(const v8bf*)p;
  v8bf hi = *(const v8bf*)(p + 16);
  return cat16(lo, hi);
}

// Async memory->LDS copy of 16B per lane (CDNA5 ASYNCcnt path).
__device__ __forceinline__ void async_load_b128(unsigned int lds_off, const void* g) {
  asm volatile("global_load_async_to_lds_b128 %0, %1, off"
               :: "v"(lds_off), "v"((unsigned long long)(uintptr_t)g)
               : "memory");
}
__device__ __forceinline__ void wait_async0() {
  asm volatile("s_wait_asynccnt 0x0" ::: "memory");
}

// ---------------------------------------------------------------------------
// Kernel 1: qkv = x(8192x1024) @ Wqkv(1024x3072) + bqkv -> q/k/v [B][H][L][D]
// 256 threads (8 waves). Block tile 256(M) x 64(N), K-step 32; each wave owns
// a 32-row strip: 2 A-frags x 4 B-frags = 8 wmma per step (1.5 b128/wmma).
// ---------------------------------------------------------------------------
__global__ __launch_bounds__(256) void qkv_gemm(const float* __restrict__ x,
                                                const float* __restrict__ w,
                                                const float* __restrict__ bias,
                                                __bf16* __restrict__ qb,
                                                __bf16* __restrict__ kb,
                                                __bf16* __restrict__ vb) {
  __shared__ __bf16 As[256 * 40];  // 256 x 32 (ld 40 -> 80B rows, 16B aligned)
  __shared__ __bf16 Bt[64 * 40];   // transposed: 64 n x 32 k
  const int tid = threadIdx.x, wave = tid >> 5, lane = tid & 31;
  const int half = lane >> 4, r = lane & 15;
  const int m_base = blockIdx.y * 256, n_base = blockIdx.x * 64;

  v8f accA[4] = {v8f{}, v8f{}, v8f{}, v8f{}};
  v8f accB[4] = {v8f{}, v8f{}, v8f{}, v8f{}};

  for (int k0 = 0; k0 < Cn; k0 += 32) {
    __syncthreads();
    // A tile: 2048 float4 chunks, 8 per thread (2 groups of 4 to cap regs).
#pragma unroll
    for (int grp = 0; grp < 2; ++grp) {
      float4 fa[4];
#pragma unroll
      for (int it = 0; it < 4; ++it) {
        int i = tid + (grp * 4 + it) * 256;
        int rr = i >> 3, cc = (i & 7) * 4;
        fa[it] = *(const float4*)&x[(long)(m_base + rr) * Cn + k0 + cc];
      }
#pragma unroll
      for (int it = 0; it < 4; ++it) {
        int i = tid + (grp * 4 + it) * 256;
        int rr = i >> 3, cc = (i & 7) * 4;
        v4bf o;
        o[0] = (__bf16)fa[it].x; o[1] = (__bf16)fa[it].y;
        o[2] = (__bf16)fa[it].z; o[3] = (__bf16)fa[it].w;
        *(v4bf*)&As[rr * 40 + cc] = o;
      }
    }
    // B tile transposed: column-wise loads (4 coalesced b32 from consecutive
    // rows per lane) -> packed v4bf -> one ds_store_b64. 512 tasks, 2/thread.
#pragma unroll
    for (int it = 0; it < 2; ++it) {
      int i = tid + it * 256;
      int col = i & 63, rc = (i >> 6) * 4;
      const float* wp = &w[(long)(k0 + rc) * (3 * Cn) + n_base + col];
      float w0 = wp[0], w1 = wp[3 * Cn], w2 = wp[6 * Cn], w3 = wp[9 * Cn];
      v4bf o;
      o[0] = (__bf16)w0; o[1] = (__bf16)w1; o[2] = (__bf16)w2; o[3] = (__bf16)w3;
      *(v4bf*)&Bt[col * 40 + rc] = o;
    }
    if (k0 + 32 < Cn) {
      __builtin_prefetch(&x[(long)(m_base + tid) * Cn + k0 + 32], 0, 1);
      __builtin_prefetch(&w[(long)(k0 + 32 + (tid >> 3)) * (3 * Cn) + n_base], 0, 1);
    }
    __syncthreads();
    // Load all fragments first, then issue the wmma chain.
    v16bf af0 = load_a_frag(As, wave * 32, 0, 40);
    v16bf af1 = load_a_frag(As, wave * 32 + 16, 0, 40);
    v16bf bf[4];
#pragma unroll
    for (int t = 0; t < 4; ++t) bf[t] = load_b_frag_nk(Bt, t * 16, 0, 40);
#pragma unroll
    for (int t = 0; t < 4; ++t) accA[t] = wmma_bf16(af0, bf[t], accA[t]);
#pragma unroll
    for (int t = 0; t < 4; ++t) accB[t] = wmma_bf16(af1, bf[t], accB[t]);
  }

#pragma unroll
  for (int t = 0; t < 4; ++t) {
    const int col = n_base + t * 16 + r;
    const float bv = bias[col];
    const int which = col >> 10;  // 0=q 1=k 2=v
    const int cc = col & (Cn - 1);
    const int hh = cc >> 6, dd = cc & 63;
    __bf16* dst = (which == 0) ? qb : ((which == 1) ? kb : vb);
#pragma unroll
    for (int g = 0; g < 2; ++g) {
#pragma unroll
      for (int j = 0; j < 8; ++j) {
        const int row = m_base + wave * 32 + g * 16 + j + 8 * half;
        const int bb = row >> 11, ll = row & (Ln - 1);
        const float v = (g ? accB[t][j] : accA[t][j]) + bv;
        dst[(((long)(bb * Hn + hh) * Ln) + ll) * Dn + dd] = (__bf16)v;
      }
    }
  }
}

// ---------------------------------------------------------------------------
// Kernel 2: per-head RMS norm of q and k in place (one wave per row of 64).
// ---------------------------------------------------------------------------
__global__ __launch_bounds__(256) void rms_kernel(const float* __restrict__ qg,
                                                  const float* __restrict__ kg,
                                                  __bf16* __restrict__ qb,
                                                  __bf16* __restrict__ kb) {
  const int wave = threadIdx.x >> 5, lane = threadIdx.x & 31;
  const long BHL = (long)Bn * Hn * Ln;
  long row = (long)blockIdx.x * 8 + wave;  // 0 .. 2*BHL-1
  const int which = (row >= BHL) ? 1 : 0;
  long rem = which ? row - BHL : row;
  const int h = (int)((rem / Ln) & (Hn - 1));
  __bf16* p = (which ? kb : qb) + rem * Dn;
  const float* g = (which ? kg : qg) + h * Dn;

  v2bf xv = *(const v2bf*)(p + lane * 2);
  float x0 = (float)xv[0], x1 = (float)xv[1];
  float ss = x0 * x0 + x1 * x1;
#pragma unroll
  for (int m = 16; m >= 1; m >>= 1) ss += __shfl_xor(ss, m, 32);
  const float nrm = sqrtf(ss);
  const float s = 8.0f / fmaxf(nrm, 1e-12f);  // sqrt(D)=8
  v2bf o;
  o[0] = (__bf16)(x0 * s * g[lane * 2]);
  o[1] = (__bf16)(x1 * s * g[lane * 2 + 1]);
  *(v2bf*)(p + lane * 2) = o;
}

// ---------------------------------------------------------------------------
// Kernel 3: flash attention, fixed max-shift. Grid (L/128, B*H), 256 threads
// (8 waves x 16 query rows). Key blocks of 32. Q/K via async LDS DMA; V
// stored transposed with permuted key columns matching packed P stores.
// ---------------------------------------------------------------------------
__global__ __launch_bounds__(256) void attn_kernel(const __bf16* __restrict__ qbuf,
                                                   const __bf16* __restrict__ kbuf,
                                                   const __bf16* __restrict__ vbuf,
                                                   __bf16* __restrict__ hb) {
  __shared__ __bf16 qs[128 * 72];     // 128 q rows x 64 d (ld 72)
  __shared__ __bf16 ks[32 * 72];      // 32 keys x 64 d (ld 72)
  __shared__ __bf16 vt[64 * 40];      // transposed V: 64 d x 32 keys' (ld 40)
  __shared__ __bf16 ps[8 * 16 * 32];  // per-wave P staging (16 x 32)

  const int tid = threadIdx.x, wave = tid >> 5, lane = tid & 31;
  const int half = lane >> 4, r = lane & 15;
  const int bh = blockIdx.y, b = bh >> 4, h = bh & (Hn - 1);
  const int q0 = blockIdx.x * 128;

  const __bf16* Q = qbuf + ((long)bh * Ln + q0) * Dn;
  const __bf16* K = kbuf + (long)bh * Ln * Dn;
  const __bf16* V = vbuf + (long)bh * Ln * Dn;

  // Q tile (128x64): async DMA, 1024 x 16B chunks, 4 per thread.
#pragma unroll
  for (int it = 0; it < 4; ++it) {
    int i = tid + it * 256;
    int rr = i >> 3, cc = (i & 7) * 8;
    async_load_b128((unsigned int)(uintptr_t)&qs[rr * 72 + cc], &Q[rr * Dn + cc]);
  }
  wait_async0();
  __syncthreads();
  const v16bf qlo = load_a_frag(qs, wave * 16, 0, 72);
  const v16bf qhi = load_a_frag(qs, wave * 16, 32, 72);

  float lsum[8];
  v8f o0 = {}, o1 = {}, o2 = {}, o3 = {};
#pragma unroll
  for (int j = 0; j < 8; ++j) lsum[j] = 0.0f;

  for (int kb0 = 0; kb0 < Ln; kb0 += 32) {
    __syncthreads();
    // K tile: async DMA (256 chunks, 1 per thread).
    {
      int rr = tid >> 3, cc = (tid & 7) * 8;
      async_load_b128((unsigned int)(uintptr_t)&ks[rr * 72 + cc],
                      &K[(kb0 + rr) * Dn + cc]);
    }
    // V tile: b128 load, transposed store with permuted key column:
    // key rr -> column (rr<16 ? 2rr : 2(rr-16)+1), matching packed P stores.
    {
      int rr = tid >> 3, cc = (tid & 7) * 8;
      int col = (rr < 16) ? (2 * rr) : (2 * (rr - 16) + 1);
      v8bf f = *(const v8bf*)&V[(kb0 + rr) * Dn + cc];
#pragma unroll
      for (int e = 0; e < 8; ++e) vt[(cc + e) * 40 + col] = f[e];
    }
    wait_async0();
    __syncthreads();

    // Load all K/V fragments up front, then compute.
    v16bf kf[4], vf[4];
    kf[0] = load_b_frag_nk(ks, 0, 0, 72);
    kf[1] = load_b_frag_nk(ks, 0, 32, 72);
    kf[2] = load_b_frag_nk(ks, 16, 0, 72);
    kf[3] = load_b_frag_nk(ks, 16, 32, 72);
#pragma unroll
    for (int t = 0; t < 4; ++t) vf[t] = load_b_frag_nk(vt, t * 16, 0, 40);

    // S = Q (16x64) @ K^T (64x32): two key sub-tiles, chained over d.
    v8f s0 = {}, s1 = {};
    s0 = wmma_bf16(qlo, kf[0], s0);
    s0 = wmma_bf16(qhi, kf[1], s0);
    s1 = wmma_bf16(qlo, kf[2], s1);
    s1 = wmma_bf16(qhi, kf[3], s1);

    // Fixed-shift softmax numerator; per-lane sum accumulation (no shuffles).
    __bf16* pw = ps + wave * 512;
#pragma unroll
    for (int j = 0; j < 8; ++j) {
      const float p0 = __expf(s0[j] * 0.125f - 8.0f);
      const float p1 = __expf(s1[j] * 0.125f - 8.0f);
      lsum[j] += p0 + p1;
      v2bf pp; pp[0] = (__bf16)p0; pp[1] = (__bf16)p1;
      *(v2bf*)&pw[(j + 8 * half) * 32 + 2 * r] = pp;  // one b32 store
    }

    // O += P (16x32) @ V (32x64) with permuted-key transposed V.
    const v16bf pf = load_a_frag(pw, 0, 0, 32);
    o0 = wmma_bf16(pf, vf[0], o0);
    o1 = wmma_bf16(pf, vf[1], o1);
    o2 = wmma_bf16(pf, vf[2], o2);
    o3 = wmma_bf16(pf, vf[3], o3);
  }

#pragma unroll
  for (int j = 0; j < 8; ++j) {
    float s = lsum[j];
#pragma unroll
    for (int m = 8; m >= 1; m >>= 1) s += __shfl_xor(s, m, 32);  // once, at end
    const float inv = 1.0f / s;
    const int rowg = q0 + wave * 16 + j + 8 * half;
    const long base = ((long)b * Ln + rowg) * Cn + h * Dn;
    hb[base + r]      = (__bf16)(o0[j] * inv);
    hb[base + 16 + r] = (__bf16)(o1[j] * inv);
    hb[base + 32 + r] = (__bf16)(o2[j] * inv);
    hb[base + 48 + r] = (__bf16)(o3[j] * inv);
  }
}

// ---------------------------------------------------------------------------
// Kernel 4: out = h(8192x1024,bf16) @ Wout(1024x1024) + bout  (fp32 out)
// Same 256x64 tiling as kernel 1.
// ---------------------------------------------------------------------------
__global__ __launch_bounds__(256) void out_gemm(const __bf16* __restrict__ hbuf,
                                                const float* __restrict__ w,
                                                const float* __restrict__ bias,
                                                float* __restrict__ out) {
  __shared__ __bf16 As[256 * 40];
  __shared__ __bf16 Bt[64 * 40];
  const int tid = threadIdx.x, wave = tid >> 5, lane = tid & 31;
  const int half = lane >> 4, r = lane & 15;
  const int m_base = blockIdx.y * 256, n_base = blockIdx.x * 64;

  v8f accA[4] = {v8f{}, v8f{}, v8f{}, v8f{}};
  v8f accB[4] = {v8f{}, v8f{}, v8f{}, v8f{}};

  for (int k0 = 0; k0 < Cn; k0 += 32) {
    __syncthreads();
    // A tile: 1024 b128 chunks of bf16, 4 per thread.
    v8bf ha[4];
#pragma unroll
    for (int it = 0; it < 4; ++it) {
      int i = tid + it * 256;
      int rr = i >> 2, cc = (i & 3) * 8;
      ha[it] = *(const v8bf*)&hbuf[(long)(m_base + rr) * Cn + k0 + cc];
    }
#pragma unroll
    for (int it = 0; it < 4; ++it) {
      int i = tid + it * 256;
      int rr = i >> 2, cc = (i & 3) * 8;
      *(v8bf*)&As[rr * 40 + cc] = ha[it];
    }
    // B tile transposed via column-wise loads + packed b64 stores.
#pragma unroll
    for (int it = 0; it < 2; ++it) {
      int i = tid + it * 256;
      int col = i & 63, rc = (i >> 6) * 4;
      const float* wp = &w[(long)(k0 + rc) * Cn + n_base + col];
      float w0 = wp[0], w1 = wp[Cn], w2 = wp[2 * Cn], w3 = wp[3 * Cn];
      v4bf o;
      o[0] = (__bf16)w0; o[1] = (__bf16)w1; o[2] = (__bf16)w2; o[3] = (__bf16)w3;
      *(v4bf*)&Bt[col * 40 + rc] = o;
    }
    if (k0 + 32 < Cn) {
      __builtin_prefetch(&hbuf[(long)(m_base + tid) * Cn + k0 + 32], 0, 1);
      __builtin_prefetch(&w[(long)(k0 + 32 + (tid >> 3)) * Cn + n_base], 0, 1);
    }
    __syncthreads();
    v16bf af0 = load_a_frag(As, wave * 32, 0, 40);
    v16bf af1 = load_a_frag(As, wave * 32 + 16, 0, 40);
    v16bf bf[4];
#pragma unroll
    for (int t = 0; t < 4; ++t) bf[t] = load_b_frag_nk(Bt, t * 16, 0, 40);
#pragma unroll
    for (int t = 0; t < 4; ++t) accA[t] = wmma_bf16(af0, bf[t], accA[t]);
#pragma unroll
    for (int t = 0; t < 4; ++t) accB[t] = wmma_bf16(af1, bf[t], accB[t]);
  }

#pragma unroll
  for (int t = 0; t < 4; ++t) {
    const int col = n_base + t * 16 + r;
    const float bv = bias[col];
#pragma unroll
    for (int g = 0; g < 2; ++g) {
#pragma unroll
      for (int j = 0; j < 8; ++j) {
        const int row = m_base + wave * 32 + g * 16 + j + 8 * half;
        out[(long)row * Cn + col] = (g ? accB[t][j] : accA[t][j]) + bv;
      }
    }
  }
}

// ---------------------------------------------------------------------------
extern "C" void kernel_launch(void* const* d_in, const int* in_sizes, int n_in,
                              void* d_out, int out_size, void* d_ws, size_t ws_size,
                              hipStream_t stream) {
  (void)in_sizes; (void)n_in; (void)out_size; (void)ws_size;
  const float* x    = (const float*)d_in[0];
  const float* Wqkv = (const float*)d_in[1];
  const float* bqkv = (const float*)d_in[2];
  const float* qg   = (const float*)d_in[3];
  const float* kg   = (const float*)d_in[4];
  const float* Wout = (const float*)d_in[5];
  const float* bout = (const float*)d_in[6];
  float* out = (float*)d_out;

  const size_t SZ = (size_t)Bn * Hn * Ln * Dn;  // 8,388,608 elems
  __bf16* qb = (__bf16*)d_ws;
  __bf16* kb = qb + SZ;
  __bf16* vb = kb + SZ;
  __bf16* hb = vb + SZ;  // [B][L][C] bf16, also SZ elems

  qkv_gemm<<<dim3(3 * Cn / 64, (Bn * Ln) / 256), 256, 0, stream>>>(x, Wqkv, bqkv, qb, kb, vb);
  rms_kernel<<<dim3((2 * Bn * Hn * Ln) / 8), 256, 0, stream>>>(qg, kg, qb, kb);
  attn_kernel<<<dim3(Ln / 128, Bn * Hn), 256, 0, stream>>>(qb, kb, vb, hb);
  out_gemm<<<dim3(Cn / 64, (Bn * Ln) / 256), 256, 0, stream>>>(hb, Wout, bout, out);
}
